// Pointnet2Encoder_68427418960109
// MI455X (gfx1250) — compile-verified, matrix operand order checked by reading, not run on hardware
//
#include <hip/hip_runtime.h>

typedef __attribute__((ext_vector_type(16))) _Float16 v16h;
typedef __attribute__((ext_vector_type(8)))  _Float16 v8h;
typedef __attribute__((ext_vector_type(8)))  float    v8f;

#define Bc  16
#define Nc  2048
#define Kc  64
#define S1c 1024
#define S2c 256

// ---------------- WMMA fragment helpers (16x16x32 f16 -> f32) ----------------
// A 16x32 f16 layout: lane L holds row M=L&15; half h maps to
//   K = (h&7) + 8*( 2*(h>>3) + (L>>4) )
// => per lane the 16 halves are two contiguous 8-half chunks at
//   k0 = kt*32 + 8*(L>>4)   and   k0 + 16.
__device__ __forceinline__ int kmap(int h, int lane) {
  return (h & 7) + (((h >> 3) << 1) + (lane >> 4)) * 8;
}

__device__ __forceinline__ v16h combine16(v8h c0, v8h c1) {
  return __builtin_shufflevector(c0, c1, 0, 1, 2, 3, 4, 5, 6, 7,
                                 8, 9, 10, 11, 12, 13, 14, 15);
}

// A fragment from LDS (row-major activations, ld = multiple of 8 halves)
__device__ __forceinline__ v16h load_a(const _Float16* src, int ld, int kt, int lane) {
  const _Float16* base = src + (lane & 15) * ld + kt * 32 + ((lane >> 4) << 3);
  v8h c0 = *(const v8h*)base;
  v8h c1 = *(const v8h*)(base + 16);
  return combine16(c0, c1);
}

// B fragment from pre-shuffled weights: tile-major [ (kt*NT+nt) ][ lane ][ 16 ]
__device__ __forceinline__ v16h load_b(const _Float16* __restrict__ Wt, int NT,
                                       int nt, int kt, int lane) {
  const _Float16* base = Wt + ((((size_t)(kt * NT + nt) << 5) + lane) << 4);
  v8h c0 = *(const v8h*)base;
  v8h c1 = *(const v8h*)(base + 8);
  return combine16(c0, c1);
}

__device__ __forceinline__ v8f wmma_f16(v16h a, v16h b, v8f c) {
  return __builtin_amdgcn_wmma_f32_16x16x32_f16(false, a, false, b, (short)0, c, false, false);
}

// D element i of lane L is (M = i + 8*(L>>4), N = L&15)
__device__ __forceinline__ void store_relu_lds(_Float16* dst, int ld, int nt, v8f d,
                                               const float* __restrict__ bias, int lane) {
  int n   = nt * 16 + (lane & 15);
  float bi = bias[n];
  int mhi = (lane >> 4) * 8;
  #pragma unroll
  for (int i = 0; i < 8; ++i) {
    float v = d[i] + bi;
    dst[(mhi + i) * ld + n] = (_Float16)(v > 0.f ? v : 0.f);
  }
}

__device__ __forceinline__ void max_agg(float* buf, int nt, v8f d, const float* __restrict__ bias,
                                        int rowbase, int cnt, int lane) {
  int n   = nt * 16 + (lane & 15);
  float bi = bias[n];
  float mx = -__builtin_inff();
  int mhi = (lane >> 4) * 8;
  #pragma unroll
  for (int i = 0; i < 8; ++i) {
    int gm = rowbase + mhi + i;
    float v = d[i] + bi;
    if (gm < cnt) mx = fmaxf(mx, v);
  }
  float o = __shfl_xor(mx, 16);     // combine rows 0-7 with rows 8-15
  mx = fmaxf(mx, o);
  if (lane < 16) buf[n] = fmaxf(buf[n], mx);
}

// --------- weight shuffle: f32 [Kin x N] -> f16 fragment order ---------------
// dst[((kt*NT+nt)*32 + lane)*16 + h] = W[(kt*32+kmap(h,lane)) * N + nt*16+(lane&15)]
__global__ void shuffle_weights(const float* __restrict__ src, _Float16* __restrict__ dst,
                                int Kin, int N, int NT, int total) {
  int i = blockIdx.x * 256 + threadIdx.x;
  if (i >= total) return;
  int h = i & 15;
  int l = (i >> 4) & 31;
  int tile = i >> 9;
  int kt = tile / NT, nt = tile % NT;
  int k = kt * 32 + kmap(h, l);
  int n = nt * 16 + (l & 15);
  dst[i] = (k < Kin) ? (_Float16)src[k * N + n] : (_Float16)0.f;
}

// ---------------- farthest point sampling (one block per cloud) --------------
__global__ void __launch_bounds__(256) fps_kernel(const float* __restrict__ pos,
                                                  int N, int S, float* __restrict__ pos_out) {
  __shared__ float mind[2048];
  __shared__ float rv[256];
  __shared__ int   ri[256];
  __shared__ float lp[3];
  __shared__ int   lasts;
  int b = blockIdx.x, t = threadIdx.x;
  const float* P = pos + (size_t)b * N * 3;
  for (int i = t; i < N; i += 256) mind[i] = __builtin_inff();
  if (t == 0) lasts = 0;
  __syncthreads();
  for (int s = 0; s < S; ++s) {
    int last = lasts;
    if (t == 0) {
      lp[0] = P[last * 3 + 0]; lp[1] = P[last * 3 + 1]; lp[2] = P[last * 3 + 2];
      float* po = pos_out + (size_t)(b * S + s) * 3;
      po[0] = lp[0]; po[1] = lp[1]; po[2] = lp[2];
    }
    __syncthreads();
    float lx = lp[0], ly = lp[1], lz = lp[2];
    float bv = -1.f; int bi = 0;
    for (int i = t; i < N; i += 256) {
      float dx = P[i * 3 + 0] - lx, dy = P[i * 3 + 1] - ly, dz = P[i * 3 + 2] - lz;
      float dd = dx * dx + dy * dy + dz * dz;
      float m = fminf(mind[i], dd);
      mind[i] = m;
      if (m > bv) { bv = m; bi = i; }
    }
    rv[t] = bv; ri[t] = bi;
    __syncthreads();
    for (int off = 128; off > 0; off >>= 1) {
      if (t < off) {
        if (rv[t + off] > rv[t] || (rv[t + off] == rv[t] && ri[t + off] < ri[t])) {
          rv[t] = rv[t + off]; ri[t] = ri[t + off];
        }
      }
      __syncthreads();
    }
    if (t == 0) lasts = ri[0];
    __syncthreads();
  }
}

// ---------------- radius neighbors: K nearest within r^2 (wave per query) ----
__global__ void __launch_bounds__(128) radius_kernel(const float* __restrict__ pall,
                                                     const float* __restrict__ pq,
                                                     int N, int S, float r2,
                                                     int* __restrict__ nbr, int* __restrict__ cnt) {
  __shared__ float distbuf[4][2048];
  int wave = threadIdx.x >> 5, lane = threadIdx.x & 31;
  int q = blockIdx.x * 4 + wave;
  int b = q / S;
  const float* P = pall + (size_t)b * N * 3;
  float qx = pq[q * 3 + 0], qy = pq[q * 3 + 1], qz = pq[q * 3 + 2];
  float* d = distbuf[wave];
  for (int i = lane; i < N; i += 32) {
    float dx = P[i * 3 + 0] - qx, dy = P[i * 3 + 1] - qy, dz = P[i * 3 + 2] - qz;
    float dd = dx * dx + dy * dy + dz * dz;
    d[i] = (dd <= r2) ? dd : __builtin_inff();
  }
  __syncthreads();
  int c = 0;
  for (int j = 0; j < Kc; ++j) {
    float bv = __builtin_inff(); int bi = -1;
    for (int i = lane; i < N; i += 32) {
      float v = d[i];
      if (v < bv || (v == bv && (unsigned)i < (unsigned)bi)) { bv = v; bi = i; }
    }
    #pragma unroll
    for (int off = 16; off > 0; off >>= 1) {
      float ov = __shfl_xor(bv, off);
      int   oi = __shfl_xor(bi, off);
      if (ov < bv || (ov == bv && (unsigned)oi < (unsigned)bi)) { bv = ov; bi = oi; }
    }
    bool valid = bv < __builtin_inff();
    if (lane == 0) nbr[(size_t)q * Kc + j] = valid ? bi : 0;
    if (valid) {
      c++;
      if (lane == 0) d[bi] = __builtin_inff();
    }
    __syncthreads();
  }
  if (lane == 0) cnt[q] = c;
}

// ---------------- SA1: rel-pos -> 64 -> 64 -> 128, masked max over K --------
__global__ void __launch_bounds__(256) sa1_kernel(
    const float* __restrict__ pos, const float* __restrict__ pos1,
    const int* __restrict__ nbr, const int* __restrict__ cnt,
    const _Float16* __restrict__ W1, const float* __restrict__ B1,
    const _Float16* __restrict__ W2, const float* __restrict__ B2,
    const _Float16* __restrict__ W3, const float* __restrict__ B3,
    _Float16* __restrict__ x1) {
  __shared__ __align__(16) _Float16 h1s[8][16 * 64];
  __shared__ __align__(16) _Float16 h2s[8][16 * 64];
  __shared__ float xs[8][128];
  __builtin_prefetch(W1, 0, 1);
  __builtin_prefetch(W2, 0, 1);
  __builtin_prefetch(W3, 0, 1);
  int wave = threadIdx.x >> 5, lane = threadIdx.x & 31;
  int q = blockIdx.x * 8 + wave;
  int b = q >> 10;                       // S1 = 1024
  const float* P = pos + (size_t)b * Nc * 3;
  float qx = pos1[q * 3 + 0], qy = pos1[q * 3 + 1], qz = pos1[q * 3 + 2];
  int c = cnt[q];
  _Float16* H1 = h1s[wave];
  _Float16* H2 = h2s[wave];
  float* X = xs[wave];
  for (int i = lane; i < 128; i += 32) X[i] = -__builtin_inff();

  for (int rt = 0; rt < 4; ++rt) {
    int row = rt * 16 + (lane & 15);
    int nb = nbr[(size_t)q * Kc + row];
    float rx = P[nb * 3 + 0] - qx, ry = P[nb * 3 + 1] - qy, rz = P[nb * 3 + 2] - qz;
    v16h a;
    #pragma unroll
    for (int h = 0; h < 16; ++h) {
      int k = kmap(h, lane);
      float v = (k == 0) ? rx : (k == 1) ? ry : (k == 2) ? rz : 0.f;
      a[h] = (_Float16)v;
    }
    for (int nt = 0; nt < 4; ++nt) {              // 3 -> 64 (Kpad 32)
      v8f acc = {};
      acc = wmma_f16(a, load_b(W1, 4, nt, 0, lane), acc);
      store_relu_lds(H1, 64, nt, acc, B1, lane);
    }
    {                                              // 64 -> 64
      v16h a0 = load_a(H1, 64, 0, lane), a1 = load_a(H1, 64, 1, lane);
      for (int nt = 0; nt < 4; ++nt) {
        v8f acc = {};
        acc = wmma_f16(a0, load_b(W2, 4, nt, 0, lane), acc);
        acc = wmma_f16(a1, load_b(W2, 4, nt, 1, lane), acc);
        store_relu_lds(H2, 64, nt, acc, B2, lane);
      }
    }
    {                                              // 64 -> 128, masked max
      v16h a0 = load_a(H2, 64, 0, lane), a1 = load_a(H2, 64, 1, lane);
      for (int nt = 0; nt < 8; ++nt) {
        v8f acc = {};
        acc = wmma_f16(a0, load_b(W3, 8, nt, 0, lane), acc);
        acc = wmma_f16(a1, load_b(W3, 8, nt, 1, lane), acc);
        max_agg(X, nt, acc, B3, rt * 16, c, lane);
      }
    }
  }
  for (int i = lane; i < 128; i += 32) x1[(size_t)q * 128 + i] = (_Float16)X[i];
}

// ---------------- SA2: [x1||rel] 131(->160) -> 128 -> 128 -> 256 ------------
__global__ void __launch_bounds__(128) sa2_kernel(
    const float* __restrict__ pos1, const float* __restrict__ pos2,
    const _Float16* __restrict__ x1,
    const int* __restrict__ nbr, const int* __restrict__ cnt,
    const _Float16* __restrict__ W1, const float* __restrict__ B1,
    const _Float16* __restrict__ W2, const float* __restrict__ B2,
    const _Float16* __restrict__ W3, const float* __restrict__ B3,
    _Float16* __restrict__ x2) {
  __shared__ __align__(16) _Float16 ains[4][16 * 160];
  __shared__ __align__(16) _Float16 h1s[4][16 * 128];
  __shared__ __align__(16) _Float16 h2s[4][16 * 128];
  __shared__ float xs[4][256];
  __builtin_prefetch(W1, 0, 1);
  __builtin_prefetch(W2, 0, 1);
  __builtin_prefetch(W3, 0, 1);
  int wave = threadIdx.x >> 5, lane = threadIdx.x & 31;
  int q = blockIdx.x * 4 + wave;
  int b = q >> 8;                        // S2 = 256
  float qx = pos2[q * 3 + 0], qy = pos2[q * 3 + 1], qz = pos2[q * 3 + 2];
  int c = cnt[q];
  _Float16* AIN = ains[wave];
  _Float16* H1 = h1s[wave];
  _Float16* H2 = h2s[wave];
  float* X = xs[wave];
  for (int i = lane; i < 256; i += 32) X[i] = -__builtin_inff();

  for (int rt = 0; rt < 4; ++rt) {
    for (int r = 0; r < 16; ++r) {
      int nb = nbr[(size_t)q * Kc + rt * 16 + r];
      size_t sp = (size_t)(b * S1c + nb);
      const v8h* xf = (const v8h*)(x1 + sp * 128);
      v8h* arow = (v8h*)(AIN + r * 160);
      if (lane < 16) {                              // 128 feature halves
        arow[lane] = xf[lane];
      } else if (lane == 16) {                      // rel xyz + pad
        float rx = pos1[sp * 3 + 0] - qx, ry = pos1[sp * 3 + 1] - qy, rz = pos1[sp * 3 + 2] - qz;
        v8h t = { (_Float16)rx, (_Float16)ry, (_Float16)rz,
                  (_Float16)0.f, (_Float16)0.f, (_Float16)0.f, (_Float16)0.f, (_Float16)0.f };
        arow[16] = t;
      } else if (lane < 20) {                       // zero pad chunks 17..19
        v8h z = {};
        arow[lane] = z;
      }
    }
    {                                              // 160 -> 128
      v16h af[5];
      #pragma unroll
      for (int kt = 0; kt < 5; ++kt) af[kt] = load_a(AIN, 160, kt, lane);
      for (int nt = 0; nt < 8; ++nt) {
        v8f acc = {};
        #pragma unroll
        for (int kt = 0; kt < 5; ++kt)
          acc = wmma_f16(af[kt], load_b(W1, 8, nt, kt, lane), acc);
        store_relu_lds(H1, 128, nt, acc, B1, lane);
      }
    }
    {                                              // 128 -> 128
      v16h af[4];
      #pragma unroll
      for (int kt = 0; kt < 4; ++kt) af[kt] = load_a(H1, 128, kt, lane);
      for (int nt = 0; nt < 8; ++nt) {
        v8f acc = {};
        #pragma unroll
        for (int kt = 0; kt < 4; ++kt)
          acc = wmma_f16(af[kt], load_b(W2, 8, nt, kt, lane), acc);
        store_relu_lds(H2, 128, nt, acc, B2, lane);
      }
    }
    {                                              // 128 -> 256, masked max
      v16h af[4];
      #pragma unroll
      for (int kt = 0; kt < 4; ++kt) af[kt] = load_a(H2, 128, kt, lane);
      for (int nt = 0; nt < 16; ++nt) {
        v8f acc = {};
        #pragma unroll
        for (int kt = 0; kt < 4; ++kt)
          acc = wmma_f16(af[kt], load_b(W3, 16, nt, kt, lane), acc);
        max_agg(X, nt, acc, B3, rt * 16, c, lane);
      }
    }
  }
  for (int i = lane; i < 256; i += 32) x2[(size_t)q * 256 + i] = (_Float16)X[i];
}

// ---------------- SA3: [x2||pos2] 259(->288) -> 256 -> 512 -> 1024, global max
__global__ void __launch_bounds__(32) sa3_kernel(
    const _Float16* __restrict__ x2, const float* __restrict__ pos2,
    const _Float16* __restrict__ W1, const float* __restrict__ B1,
    const _Float16* __restrict__ W2, const float* __restrict__ B2,
    const _Float16* __restrict__ W3, const float* __restrict__ B3,
    float* __restrict__ out) {
  __shared__ __align__(16) _Float16 AIN[16 * 288];
  __shared__ __align__(16) _Float16 H1[16 * 256];
  __shared__ __align__(16) _Float16 H2[16 * 512];
  __shared__ float G[1024];
  __builtin_prefetch(W1, 0, 1);
  __builtin_prefetch(W2, 0, 1);
  __builtin_prefetch(W3, 0, 1);
  int lane = threadIdx.x;
  int b = blockIdx.x;
  for (int i = lane; i < 1024; i += 32) G[i] = -__builtin_inff();

  for (int rt = 0; rt < 16; ++rt) {
    for (int r = 0; r < 16; ++r) {
      size_t qi = (size_t)(b * S2c + rt * 16 + r);
      const v8h* xf = (const v8h*)(x2 + qi * 256);
      v8h* arow = (v8h*)(AIN + r * 288);
      arow[lane] = xf[lane];                       // 256 feature halves (32 chunks)
      if (lane == 0) {                             // pos xyz + pad (chunk 32)
        v8h t = { (_Float16)pos2[qi * 3 + 0], (_Float16)pos2[qi * 3 + 1],
                  (_Float16)pos2[qi * 3 + 2], (_Float16)0.f,
                  (_Float16)0.f, (_Float16)0.f, (_Float16)0.f, (_Float16)0.f };
        arow[32] = t;
      } else if (lane < 4) {                       // zero chunks 33..35
        v8h z = {};
        arow[32 + lane] = z;
      }
    }
    for (int nt = 0; nt < 16; ++nt) {              // 288 -> 256
      v8f acc = {};
      for (int kt = 0; kt < 9; ++kt)
        acc = wmma_f16(load_a(AIN, 288, kt, lane), load_b(W1, 16, nt, kt, lane), acc);
      store_relu_lds(H1, 256, nt, acc, B1, lane);
    }
    for (int nt = 0; nt < 32; ++nt) {              // 256 -> 512
      v8f acc = {};
      for (int kt = 0; kt < 8; ++kt)
        acc = wmma_f16(load_a(H1, 256, kt, lane), load_b(W2, 32, nt, kt, lane), acc);
      store_relu_lds(H2, 512, nt, acc, B2, lane);
    }
    for (int nt = 0; nt < 64; ++nt) {              // 512 -> 1024, max over rows
      v8f acc = {};
      for (int kt = 0; kt < 16; ++kt)
        acc = wmma_f16(load_a(H2, 512, kt, lane), load_b(W3, 64, nt, kt, lane), acc);
      max_agg(G, nt, acc, B3, 0, 16, lane);
    }
  }
  for (int i = lane; i < 1024; i += 32) out[(size_t)b * 1024 + i] = G[i];
}

// ---------------- host launcher ---------------------------------------------
extern "C" void kernel_launch(void* const* d_in, const int* in_sizes, int n_in,
                              void* d_out, int out_size, void* d_ws, size_t ws_size,
                              hipStream_t stream) {
  (void)in_sizes; (void)n_in; (void)out_size; (void)ws_size;
  const float* pos = (const float*)d_in[0];
  const float* w[9];
  const float* bb[9];
  for (int i = 0; i < 9; ++i) {
    w[i]  = (const float*)d_in[1 + 2 * i];
    bb[i] = (const float*)d_in[2 + 2 * i];
  }
  static const int Kin[9]  = {  3,  64,  64, 131, 128, 128, 259, 256,  512 };
  static const int Kpad[9] = { 32,  64,  64, 160, 128, 128, 288, 256,  512 };
  static const int Nn[9]   = { 64,  64, 128, 128, 128, 256, 256, 512, 1024 };

  char* p = (char*)d_ws;
  auto alloc = [&](size_t bytes) -> void* {
    void* r = (void*)p;
    p += (bytes + 255) & ~(size_t)255;
    return r;
  };
  float*    pos1 = (float*)   alloc((size_t)Bc * S1c * 3 * sizeof(float));
  int*      cnt1 = (int*)     alloc((size_t)Bc * S1c * sizeof(int));
  int*      nbr1 = (int*)     alloc((size_t)Bc * S1c * Kc * sizeof(int));
  _Float16* x1   = (_Float16*)alloc((size_t)Bc * S1c * 128 * sizeof(_Float16));
  float*    pos2 = (float*)   alloc((size_t)Bc * S2c * 3 * sizeof(float));
  int*      cnt2 = (int*)     alloc((size_t)Bc * S2c * sizeof(int));
  int*      nbr2 = (int*)     alloc((size_t)Bc * S2c * Kc * sizeof(int));
  _Float16* x2   = (_Float16*)alloc((size_t)Bc * S2c * 256 * sizeof(_Float16));
  _Float16* Wh[9];
  for (int i = 0; i < 9; ++i)
    Wh[i] = (_Float16*)alloc((size_t)Kpad[i] * Nn[i] * sizeof(_Float16));

  for (int i = 0; i < 9; ++i) {
    int total = Kpad[i] * Nn[i];                  // == KT*NT*512
    shuffle_weights<<<(total + 255) / 256, 256, 0, stream>>>(w[i], Wh[i], Kin[i], Nn[i],
                                                             Nn[i] / 16, total);
  }

  fps_kernel<<<Bc, 256, 0, stream>>>(pos, Nc, S1c, pos1);
  radius_kernel<<<Bc * S1c / 4, 128, 0, stream>>>(pos, pos1, Nc, S1c, 0.2f * 0.2f, nbr1, cnt1);
  sa1_kernel<<<Bc * S1c / 8, 256, 0, stream>>>(pos, pos1, nbr1, cnt1,
                                               Wh[0], bb[0], Wh[1], bb[1], Wh[2], bb[2], x1);
  fps_kernel<<<Bc, 256, 0, stream>>>(pos1, S1c, S2c, pos2);
  radius_kernel<<<Bc * S2c / 4, 128, 0, stream>>>(pos1, pos2, S1c, S2c, 0.4f * 0.4f, nbr2, cnt2);
  sa2_kernel<<<Bc * S2c / 4, 128, 0, stream>>>(pos1, pos2, x1, nbr2, cnt2,
                                               Wh[3], bb[3], Wh[4], bb[4], Wh[5], bb[5], x2);
  sa3_kernel<<<Bc, 32, 0, stream>>>(x2, pos2,
                                    Wh[6], bb[6], Wh[7], bb[7], Wh[8], bb[8], (float*)d_out);
}
